// LAN_76690936037957
// MI455X (gfx1250) — matrix operand order repaired and checked by reading, not compiled, not run on hardware
//
#include <hip/hip_runtime.h>
#include <hip/hip_bf16.h>
#include <cstdint>

// ---------------- problem constants ----------------
#define B_    2
#define T_    512
#define DM_   512
#define H_    8
#define D_    64
#define KTOP  8
#define F_    128
#define BT_   (B_ * T_)        // 1024
#define BHT_  (B_ * H_ * T_)   // 8192
#define KLSTM 640              // F_ + DM_ (fused x|h contraction)
#define NLSTM 2048             // 4 * DM_  (i,f,g,o)

typedef __bf16 bf16_t;
typedef __attribute__((ext_vector_type(16))) __bf16 v16bf;
typedef __attribute__((ext_vector_type(8)))  __bf16 v8bf;
typedef __attribute__((ext_vector_type(8)))  float  v8f;
typedef __attribute__((ext_vector_type(4)))  float  v4f;
typedef __attribute__((ext_vector_type(4)))  unsigned int v4u;
typedef __attribute__((ext_vector_type(8)))  int v8i;
typedef __attribute__((ext_vector_type(4)))  int v4i;

#if __has_builtin(__builtin_amdgcn_tensor_load_to_lds) && __has_builtin(__builtin_amdgcn_s_wait_tensorcnt)
#define HAVE_TDM 1
#else
#define HAVE_TDM 0
#endif

__device__ __forceinline__ v16bf cat8(v8bf a, v8bf b) {
  return __builtin_shufflevector(a, b, 0,1,2,3,4,5,6,7,8,9,10,11,12,13,14,15);
}
__device__ __forceinline__ float sigmoidf_(float x) { return 1.0f / (1.0f + __expf(-x)); }
__device__ __forceinline__ float softplusf_(float x) {
  return fmaxf(x, 0.0f) + log1pf(__expf(-fabsf(x)));
}

// ---------------- elementwise f32 -> bf16 ----------------
__global__ void cvt_bf16_kernel(const float* __restrict__ in, bf16_t* __restrict__ out, int n) {
  int i = blockIdx.x * blockDim.x + threadIdx.x;
  if (i < n) out[i] = (bf16_t)in[i];
}

// ---------------- transpose + convert: WT[n*ldo + kofs + k] = W[k*N + n] ----------------
__global__ void transpose_cvt_kernel(const float* __restrict__ W, bf16_t* __restrict__ WT,
                                     int K, int N, int ldo, int kofs) {
  int i = blockIdx.x * blockDim.x + threadIdx.x;
  if (i < K * N) {
    int k = i / N, n = i % N;
    WT[(size_t)n * ldo + kofs + k] = (bf16_t)W[(size_t)k * N + n];
  }
}

// ---------------- generic bf16 GEMM: out = act(A[M,K] * B[K,N] + bias) ----------------
#define GLOADB4(dst, kbh) do {                                              \
    const size_t rbg_ = (size_t)(colW + ccol) * Kd + (kbh) + bk16;          \
    dst[0] = *(const v16bf*)(BT + rbg_);                                    \
    dst[1] = *(const v16bf*)(BT + rbg_ + (size_t)16 * Kd);                  \
    dst[2] = *(const v16bf*)(BT + rbg_ + (size_t)32 * Kd);                  \
    dst[3] = *(const v16bf*)(BT + rbg_ + (size_t)48 * Kd);                  \
  } while (0)

#define GDOWMMA4(buf) do {                                                  \
    acc[0] = __builtin_amdgcn_wmma_f32_16x16x32_bf16(false, afrag, false, buf[0], (short)0, acc[0], false, false); \
    acc[1] = __builtin_amdgcn_wmma_f32_16x16x32_bf16(false, afrag, false, buf[1], (short)0, acc[1], false, false); \
    acc[2] = __builtin_amdgcn_wmma_f32_16x16x32_bf16(false, afrag, false, buf[2], (short)0, acc[2], false, false); \
    acc[3] = __builtin_amdgcn_wmma_f32_16x16x32_bf16(false, afrag, false, buf[3], (short)0, acc[3], false, false); \
  } while (0)

__global__ __launch_bounds__(256) void gemm_bf16_kernel(
    const bf16_t* __restrict__ A, const bf16_t* __restrict__ BT,
    const float* __restrict__ bias, float* __restrict__ outF,
    int M, int N, int Kd, int act)
{
  __shared__ bf16_t Asub[64][72];
  const int tid  = threadIdx.x;
  const int lane = tid & 31;
  const int wid  = tid >> 5;
  const int wm   = wid & 3;
  const int wn   = wid >> 2;
  const int rowG = blockIdx.y * 64;
  const int rowW = wm * 16;
  const int colW = blockIdx.x * 128 + wn * 64;

  const int crow = (lane >> 4) ? 8 : 0;
  const int ccol = lane & 15;
  const int fr   = lane & 15;
  const int fk0  = (lane >> 4) ? 16 : 0;
  const int bk16 = (lane >> 4) ? 16 : 0;

  v8f acc[4];
  #pragma unroll
  for (int nt = 0; nt < 4; ++nt) {
    float b = bias[colW + nt * 16 + ccol];
    #pragma unroll
    for (int e = 0; e < 8; ++e) acc[nt][e] = b;
  }

  const int ar = tid >> 2;
  const int ac = (tid & 3) * 8;

  v16bf bb0[4], bb1[4];
  GLOADB4(bb0, 0);                            // pipeline prologue (kb=0, half 0)

  for (int kb = 0; kb < Kd; kb += 64) {
    v8bf a0 = *(const v8bf*)(A + (size_t)(rowG + ar) * Kd + kb + ac);
    v8bf a1 = *(const v8bf*)(A + (size_t)(rowG + ar) * Kd + kb + ac + 32);
    *(v8bf*)(&Asub[ar][ac])      = a0;
    *(v8bf*)(&Asub[ar][ac + 32]) = a1;
    __syncthreads();

    v16bf afrag;
    // half 0: wmma bb0, preload half 1 into bb1
    afrag = cat8(*(const v8bf*)(&Asub[rowW + fr][fk0]),
                 *(const v8bf*)(&Asub[rowW + fr][fk0 + 8]));
    GLOADB4(bb1, kb + 32);
    GDOWMMA4(bb0);
    // half 1: wmma bb1, preload next kb's half 0 into bb0 (tail over-read is padded)
    afrag = cat8(*(const v8bf*)(&Asub[rowW + fr][32 + fk0]),
                 *(const v8bf*)(&Asub[rowW + fr][32 + fk0 + 8]));
    GLOADB4(bb0, kb + 64);
    GDOWMMA4(bb1);
    __syncthreads();
  }

  #pragma unroll
  for (int nt = 0; nt < 4; ++nt) {
    #pragma unroll
    for (int e = 0; e < 8; ++e) {
      float v = acc[nt][e];
      if (act == 1) v = sigmoidf_(v);
      outF[(size_t)(rowG + rowW + crow + e) * N + colW + nt * 16 + ccol] = v;
    }
  }
}

// ---------------- scores + top-8 + pair tensor ----------------
__global__ __launch_bounds__(256) void topk_pair_kernel(
    const float* __restrict__ q, const float* __restrict__ k,
    int* __restrict__ topk_idx, bf16_t* __restrict__ pairT)
{
  const int lane = threadIdx.x & 31;
  const int wid  = threadIdx.x >> 5;
  const int row  = blockIdx.x * 8 + wid;
  const int bh = row / T_, t = row % T_;
  const int b  = bh / H_,  h = bh % H_;

  const float* qrow  = q + (size_t)(b * T_ + t) * DM_ + h * D_;
  const float* kbase = k + (size_t)b * T_ * DM_ + h * D_;

  float qv[D_];
  #pragma unroll
  for (int i = 0; i < D_ / 4; ++i) {
    v4f x = *(const v4f*)(qrow + i * 4);
    qv[i*4+0] = x[0]; qv[i*4+1] = x[1]; qv[i*4+2] = x[2]; qv[i*4+3] = x[3];
  }

  float ls[KTOP]; int li[KTOP];
  #pragma unroll
  for (int j = 0; j < KTOP; ++j) { ls[j] = -3.0e38f; li[j] = 0x7fffffff; }

  for (int it = 0; it < T_ / 32; ++it) {
    int key = it * 32 + lane;
    const float* krow = kbase + (size_t)key * DM_;
    float dot = 0.f;
    #pragma unroll
    for (int i = 0; i < D_ / 4; ++i) {
      v4f x = *(const v4f*)(krow + i * 4);
      dot += qv[i*4+0]*x[0] + qv[i*4+1]*x[1] + qv[i*4+2]*x[2] + qv[i*4+3]*x[3];
    }
    float v = dot; int id = key;
    #pragma unroll
    for (int j = 0; j < KTOP; ++j) {
      bool gt = (v > ls[j]) || (v == ls[j] && id < li[j]);
      float tv = gt ? ls[j] : v;  int ti = gt ? li[j] : id;
      ls[j] = gt ? v : ls[j];     li[j] = gt ? id : li[j];
      v = tv; id = ti;
    }
  }

  int sel[KTOP];
  #pragma unroll
  for (int r = 0; r < KTOP; ++r) {
    float v = ls[0]; int id = li[0];
    #pragma unroll
    for (int off = 16; off >= 1; off >>= 1) {
      float ov = __shfl_xor(v, off, 32);
      int   oi = __shfl_xor(id, off, 32);
      if (ov > v || (ov == v && oi < id)) { v = ov; id = oi; }
    }
    sel[r] = id;
    if (lane == 0) topk_idx[(size_t)row * KTOP + r] = id;
    if (li[0] == id) {
      #pragma unroll
      for (int j = 0; j < KTOP - 1; ++j) { ls[j] = ls[j+1]; li[j] = li[j+1]; }
      ls[KTOP-1] = -3.0e38f; li[KTOP-1] = 0x7fffffff;
    }
  }

  #pragma unroll
  for (int kk = 0; kk < KTOP; ++kk) {
    const float* krow = kbase + (size_t)sel[kk] * DM_;
    bf16_t* dst = pairT + ((size_t)kk * BHT_ + row) * F_;
    #pragma unroll
    for (int j = 0; j < F_ / 32; ++j) {
      int f = lane + j * 32;
      float v = (f < D_) ? qrow[f] : krow[f - D_];
      dst[f] = (bf16_t)v;
    }
  }
}

// ---------------- LSTM over K=8 steps (the hot loop) ----------------
#if HAVE_TDM
__device__ __forceinline__ void tdm_load_x(const bf16_t* src, unsigned ldsXs) {
  // D#: 1-D tensor, 2048 x 2B contiguous, LDS pad 8 DWORDs every 64 DWORDs.
  uint64_t ga = (uint64_t)(uintptr_t)src;
  v4u g0;
  g0[0] = 1u;
  g0[1] = ldsXs;
  g0[2] = (unsigned)(ga & 0xffffffffu);
  g0[3] = (unsigned)((ga >> 32) & 0x1ffffffu) | (2u << 30);
  v8i g1;
  g1[0] = (1 << 16) | (1 << 20) | (5 << 22) | (7 << 25);
  g1[1] = (int)(2048u << 16);
  g1[2] = (int)(1u << 16);
  g1[3] = (int)(2048u << 16);
  g1[4] = 1;
  g1[5] = 2048;
  g1[6] = 0;
  g1[7] = 0;
  v4i g2; g2[0]=0; g2[1]=0; g2[2]=0; g2[3]=0;
  v4i g3; g3[0]=0; g3[1]=0; g3[2]=0; g3[3]=0;
#if __has_include(<hip/amd_detail/amd_gfx1250_TDM.h>)
  v8i g4; g4[0]=0; g4[1]=0; g4[2]=0; g4[3]=0; g4[4]=0; g4[5]=0; g4[6]=0; g4[7]=0;
  __builtin_amdgcn_tensor_load_to_lds(g0, g1, g2, g3, g4, 0);
#else
  __builtin_amdgcn_tensor_load_to_lds(g0, g1, g2, g3, 0);
#endif
}
#endif

// load 8 B fragments (two gate-groups gA,gB x 4 hidden tiles) at K offset kc
#define LOADB8(dst, gA, gB, kc) do {                                          \
    const size_t rbA_ = (size_t)((gA) * DM_ + hb + ccol) * KLSTM + (kc) + bk16; \
    const size_t rbB_ = (size_t)((gB) * DM_ + hb + ccol) * KLSTM + (kc) + bk16; \
    dst[0] = *(const v16bf*)(WT + rbA_);                                      \
    dst[1] = *(const v16bf*)(WT + rbA_ + (size_t)16 * KLSTM);                 \
    dst[2] = *(const v16bf*)(WT + rbA_ + (size_t)32 * KLSTM);                 \
    dst[3] = *(const v16bf*)(WT + rbA_ + (size_t)48 * KLSTM);                 \
    dst[4] = *(const v16bf*)(WT + rbB_);                                      \
    dst[5] = *(const v16bf*)(WT + rbB_ + (size_t)16 * KLSTM);                 \
    dst[6] = *(const v16bf*)(WT + rbB_ + (size_t)32 * KLSTM);                 \
    dst[7] = *(const v16bf*)(WT + rbB_ + (size_t)48 * KLSTM);                 \
  } while (0)

#define DOWMMA2(gA, gB, buf) do {                                             \
    acc[gA][0] = __builtin_amdgcn_wmma_f32_16x16x32_bf16(false, afrag, false, buf[0], (short)0, acc[gA][0], false, false); \
    acc[gA][1] = __builtin_amdgcn_wmma_f32_16x16x32_bf16(false, afrag, false, buf[1], (short)0, acc[gA][1], false, false); \
    acc[gA][2] = __builtin_amdgcn_wmma_f32_16x16x32_bf16(false, afrag, false, buf[2], (short)0, acc[gA][2], false, false); \
    acc[gA][3] = __builtin_amdgcn_wmma_f32_16x16x32_bf16(false, afrag, false, buf[3], (short)0, acc[gA][3], false, false); \
    acc[gB][0] = __builtin_amdgcn_wmma_f32_16x16x32_bf16(false, afrag, false, buf[4], (short)0, acc[gB][0], false, false); \
    acc[gB][1] = __builtin_amdgcn_wmma_f32_16x16x32_bf16(false, afrag, false, buf[5], (short)0, acc[gB][1], false, false); \
    acc[gB][2] = __builtin_amdgcn_wmma_f32_16x16x32_bf16(false, afrag, false, buf[6], (short)0, acc[gB][2], false, false); \
    acc[gB][3] = __builtin_amdgcn_wmma_f32_16x16x32_bf16(false, afrag, false, buf[7], (short)0, acc[gB][3], false, false); \
  } while (0)

__global__ __launch_bounds__(256) void lstm_kernel(
    const bf16_t* __restrict__ pairT, const bf16_t* __restrict__ WT,
    const float* __restrict__ b_lstm, const float* __restrict__ Wg,
    const float* __restrict__ bg, float* __restrict__ gate_raw)
{
  __shared__ bf16_t Xs[16][144];    // x_t staging (TDM dest, padded rows)
  __shared__ bf16_t Hs[16][528];    // recurrent h (bf16), padded rows
  __shared__ float  gred[16][17];

  const int tid  = threadIdx.x;
  const int lane = tid & 31;
  const int wid  = tid >> 5;
  const int rowBase = blockIdx.x * 16;
  const int hb   = wid * 64;
  const int crow = (lane >> 4) ? 8 : 0;
  const int ccol = lane & 15;
  const int fr   = lane & 15;
  const int fk0  = (lane >> 4) ? 16 : 0;
  const int bk16 = (lane >> 4) ? 16 : 0;

  { // zero h region (16 x 512)
    v8bf z;
    #pragma unroll
    for (int i = 0; i < 8; ++i) z[i] = (bf16_t)0.0f;
    #pragma unroll
    for (int j = 0; j < 4; ++j) {
      int e = tid * 32 + j * 8;
      int r = e >> 9, c = e & 511;
      *(v8bf*)(&Hs[r][c]) = z;
    }
  }

  v8f cst[4];
  #pragma unroll
  for (int t = 0; t < 4; ++t)
    #pragma unroll
    for (int e = 0; e < 8; ++e) cst[t][e] = 0.f;

  float bias[4][4];
  #pragma unroll
  for (int g = 0; g < 4; ++g)
    #pragma unroll
    for (int t = 0; t < 4; ++t)
      bias[g][t] = b_lstm[g * DM_ + hb + t * 16 + ccol];

#if HAVE_TDM
  const unsigned ldsXs = (unsigned)(uintptr_t)&Xs[0][0];
  if (wid == 0) tdm_load_x(pairT + (size_t)rowBase * F_, ldsXs);   // step 0 x tile
#endif

  for (int kk = 0; kk < KTOP; ++kk) {
#if HAVE_TDM
    if (wid == 0) __builtin_amdgcn_s_wait_tensorcnt(0);
#else
    { // manual stage x_t (16x128 bf16)
      int r = tid >> 4, c = (tid & 15) * 8;
      v8bf xv = *(const v8bf*)(pairT + ((size_t)kk * BHT_ + rowBase + r) * F_ + c);
      *(v8bf*)(&Xs[r][c]) = xv;
    }
#endif
    __syncthreads();   // Xs visible to all waves

    v8f acc[4][4];
    #pragma unroll
    for (int g = 0; g < 4; ++g)
      #pragma unroll
      for (int t = 0; t < 4; ++t)
        #pragma unroll
        for (int e = 0; e < 8; ++e) acc[g][t][e] = bias[g][t];

    v16bf bb0[8], bb1[8];
    LOADB8(bb0, 0, 1, 0);                   // pipeline prologue

    // ---- x part: K = 0..127 from Xs (16-load-deep B double-buffer) ----
    #pragma unroll
    for (int c = 0; c < 4; ++c) {
      const int kc = c * 32;
      v16bf afrag = cat8(*(const v8bf*)(&Xs[fr][kc + fk0]),
                         *(const v8bf*)(&Xs[fr][kc + fk0 + 8]));
      LOADB8(bb1, 2, 3, kc);       DOWMMA2(0, 1, bb0);
      LOADB8(bb0, 0, 1, kc + 32);  DOWMMA2(2, 3, bb1);   // c=3 preloads h-part kc=128
    }

    __syncthreads();   // all waves finished reading Xs
#if HAVE_TDM
    if (kk < KTOP - 1 && wid == 0)
      tdm_load_x(pairT + ((size_t)(kk + 1) * BHT_ + rowBase) * F_, ldsXs);  // overlap DMA with h-part
#endif

    // ---- h part: K = 128..639 from Hs (pipelined) ----
    for (int kc = F_; kc < KLSTM; kc += 32) {
      __builtin_prefetch(WT + (size_t)(hb + ccol) * KLSTM + kc + 64, 0, 0);
      v16bf afrag = cat8(*(const v8bf*)(&Hs[fr][kc - F_ + fk0]),
                         *(const v8bf*)(&Hs[fr][kc - F_ + fk0 + 8]));
      LOADB8(bb1, 2, 3, kc);       DOWMMA2(0, 1, bb0);
      LOADB8(bb0, 0, 1, kc + 32);  DOWMMA2(2, 3, bb1);   // last iter over-reads 64B (WT padded)
    }
    __syncthreads();

    // gates + state update; write h (bf16) into Hs for next step
    #pragma unroll
    for (int t = 0; t < 4; ++t) {
      #pragma unroll
      for (int e = 0; e < 8; ++e) {
        float iF = sigmoidf_(acc[0][t][e]);
        float fF = sigmoidf_(acc[1][t][e]);
        float gF = tanhf(acc[2][t][e]);
        float oF = sigmoidf_(acc[3][t][e]);
        float c  = fF * cst[t][e] + iF * gF;
        cst[t][e] = c;
        float hV = oF * tanhf(c);
        Hs[crow + e][hb + t * 16 + ccol] = (bf16_t)hV;
      }
    }
    __syncthreads();

    { // gate = h . Wg + bg (per-row reduction through LDS)
      int r = tid >> 4, seg = tid & 15;
      float part = 0.f;
      #pragma unroll
      for (int j = 0; j < 32; ++j) {
        int hid = seg * 32 + j;
        part += (float)Hs[r][hid] * Wg[hid];
      }
      gred[r][seg] = part;
    }
    __syncthreads();
    if (tid < 16) {
      float s = bg[0];
      #pragma unroll
      for (int j = 0; j < 16; ++j) s += gred[tid][j];
      gate_raw[(size_t)(rowBase + tid) * KTOP + kk] = s;
    }
    __syncthreads();
  }
}

// ---------------- global max(tau) (single block) ----------------
__global__ __launch_bounds__(256) void taumax_kernel(const float* __restrict__ gate_raw,
                                                     float* __restrict__ taumax) {
  __shared__ float red[256];
  float m = -3.0e38f;
  for (int i = threadIdx.x; i < BHT_ * KTOP; i += 256) {
    float tau = softplusf_(gate_raw[i]) + 1e-6f;
    m = fmaxf(m, tau);
  }
  red[threadIdx.x] = m;
  __syncthreads();
  for (int s = 128; s > 0; s >>= 1) {
    if (threadIdx.x < (unsigned)s) red[threadIdx.x] = fmaxf(red[threadIdx.x], red[threadIdx.x + s]);
    __syncthreads();
  }
  if (threadIdx.x == 0) taumax[0] = red[0];
}

// ---------------- Euler + softmax + top-k value gather + sink gate ----------------
__global__ __launch_bounds__(256) void attn_out_kernel(
    const float* __restrict__ gate_raw, const float* __restrict__ taumax,
    const int* __restrict__ topk_idx, const float* __restrict__ v,
    const float* __restrict__ sink, bf16_t* __restrict__ combined)
{
  const int lane = threadIdx.x & 31;
  const int wid  = threadIdx.x >> 5;
  const int row  = blockIdx.x * 8 + wid;
  const int bh = row / T_, t = row % T_;
  const int b  = bh / H_,  h = bh % H_;

  const float dt = fminf(0.01f, 1.0f / (taumax[0] + 1e-12f));

  float a[KTOP];
  float mx = -3.0e38f;
  #pragma unroll
  for (int j = 0; j < KTOP; ++j) {
    float x   = gate_raw[(size_t)row * KTOP + j];
    float phi = fmaxf(x, 0.f);
    float tau = softplusf_(x) + 1e-6f;
    float aa = 0.f;
    #pragma unroll
    for (int s = 0; s < 2; ++s) aa = aa + dt * (-tau * aa + phi);
    a[j] = aa;
    mx = fmaxf(mx, aa);
  }
  float ssum = 0.f;
  #pragma unroll
  for (int j = 0; j < KTOP; ++j) { a[j] = __expf(a[j] - mx); ssum += a[j]; }
  const float inv = 1.0f / ssum;

  int idx[KTOP];
  #pragma unroll
  for (int j = 0; j < KTOP; ++j) idx[j] = topk_idx[(size_t)row * KTOP + j];

  #pragma unroll
  for (int jd = 0; jd < D_ / 32; ++jd) {
    int d = lane + jd * 32;
    float o = 0.f;
    #pragma unroll
    for (int j = 0; j < KTOP; ++j)
      o += a[j] * inv * v[(size_t)(b * T_ + idx[j]) * DM_ + h * D_ + d];
    size_t oix = (size_t)(b * T_ + t) * DM_ + h * D_ + d;
    combined[oix] = (bf16_t)(o * sink[oix]);
  }
}

// ---------------- host orchestration ----------------
extern "C" void kernel_launch(void* const* d_in, const int* in_sizes, int n_in,
                              void* d_out, int out_size, void* d_ws, size_t ws_size,
                              hipStream_t stream)
{
  const float* x  = (const float*)d_in[0];
  const float* Wq = (const float*)d_in[1];
  const float* bq = (const float*)d_in[2];
  const float* Wk = (const float*)d_in[3];
  const float* bk = (const float*)d_in[4];
  const float* Wv = (const float*)d_in[5];
  const float* bv = (const float*)d_in[6];
  const float* Wl = (const float*)d_in[7];
  const float* Ul = (const float*)d_in[8];
  const float* bl = (const float*)d_in[9];
  const float* Wg = (const float*)d_in[10];
  const float* bg = (const float*)d_in[11];
  const float* Ws = (const float*)d_in[12];
  const float* bs = (const float*)d_in[13];
  const float* Wo = (const float*)d_in[14];
  const float* bo = (const float*)d_in[15];

  char* w = (char*)d_ws;
  size_t off = 0;
  auto alloc = [&](size_t bytes) -> char* {
    char* p = w + off;
    off = (off + bytes + 255) & ~(size_t)255;
    return p;
  };

  bf16_t* xb   = (bf16_t*)alloc((size_t)BT_ * DM_ * 2);
  bf16_t* WqT  = (bf16_t*)alloc((size_t)DM_ * DM_ * 2);
  bf16_t* WkT  = (bf16_t*)alloc((size_t)DM_ * DM_ * 2);
  bf16_t* WvT  = (bf16_t*)alloc((size_t)DM_ * DM_ * 2);
  bf16_t* WsT  = (bf16_t*)alloc((size_t)DM_ * DM_ * 2);
  bf16_t* WoT  = (bf16_t*)alloc((size_t)DM_ * DM_ * 2);
  bf16_t* WTl  = (bf16_t*)alloc((size_t)NLSTM * KLSTM * 2 + 256);  // +pad for pipelined over-read
  float*  qf   = (float*)alloc((size_t)BT_ * DM_ * 4);
  float*  kf   = (float*)alloc((size_t)BT_ * DM_ * 4);
  float*  vf   = (float*)alloc((size_t)BT_ * DM_ * 4);
  float*  sink = (float*)alloc((size_t)BT_ * DM_ * 4);
  bf16_t* pairT = (bf16_t*)alloc((size_t)KTOP * BHT_ * F_ * 2);
  int*    tidx = (int*)alloc((size_t)BHT_ * KTOP * 4);
  float*  grw  = (float*)alloc((size_t)BHT_ * KTOP * 4);
  float*  tmax = (float*)alloc(256);
  bf16_t* comb = (bf16_t*)alloc((size_t)BT_ * DM_ * 2);
  (void)ws_size; (void)n_in; (void)in_sizes; (void)out_size;

  cvt_bf16_kernel<<<(BT_ * DM_ + 255) / 256, 256, 0, stream>>>(x, xb, BT_ * DM_);
  transpose_cvt_kernel<<<(DM_ * DM_ + 255) / 256, 256, 0, stream>>>(Wq, WqT, DM_, DM_, DM_, 0);
  transpose_cvt_kernel<<<(DM_ * DM_ + 255) / 256, 256, 0, stream>>>(Wk, WkT, DM_, DM_, DM_, 0);
  transpose_cvt_kernel<<<(DM_ * DM_ + 255) / 256, 256, 0, stream>>>(Wv, WvT, DM_, DM_, DM_, 0);
  transpose_cvt_kernel<<<(DM_ * DM_ + 255) / 256, 256, 0, stream>>>(Ws, WsT, DM_, DM_, DM_, 0);
  transpose_cvt_kernel<<<(DM_ * DM_ + 255) / 256, 256, 0, stream>>>(Wo, WoT, DM_, DM_, DM_, 0);
  transpose_cvt_kernel<<<(F_ * NLSTM + 255) / 256, 256, 0, stream>>>(Wl, WTl, F_, NLSTM, KLSTM, 0);
  transpose_cvt_kernel<<<(DM_ * NLSTM + 255) / 256, 256, 0, stream>>>(Ul, WTl, DM_, NLSTM, KLSTM, F_);

  dim3 gg(DM_ / 128, BT_ / 64);
  gemm_bf16_kernel<<<gg, 256, 0, stream>>>(xb, WqT, bq, qf,   BT_, DM_, DM_, 0);
  gemm_bf16_kernel<<<gg, 256, 0, stream>>>(xb, WkT, bk, kf,   BT_, DM_, DM_, 0);
  gemm_bf16_kernel<<<gg, 256, 0, stream>>>(xb, WvT, bv, vf,   BT_, DM_, DM_, 0);
  gemm_bf16_kernel<<<gg, 256, 0, stream>>>(xb, WsT, bs, sink, BT_, DM_, DM_, 1);

  topk_pair_kernel<<<BHT_ / 8, 256, 0, stream>>>(qf, kf, tidx, pairT);

  lstm_kernel<<<BHT_ / 16, 256, 0, stream>>>(pairT, WTl, bl, Wg, bg, grw);

  taumax_kernel<<<1, 256, 0, stream>>>(grw, tmax);
  attn_out_kernel<<<BHT_ / 8, 256, 0, stream>>>(grw, tmax, tidx, vf, sink, comb);

  gemm_bf16_kernel<<<gg, 256, 0, stream>>>(comb, WoT, bo, (float*)d_out, BT_, DM_, DM_, 0);
}
